// RNNModel_40836549050807
// MI455X (gfx1250) — compile-verified, matrix operand order checked by reading, not run on hardware
//
#include <hip/hip_runtime.h>

typedef __attribute__((ext_vector_type(16))) __bf16 v16bf;
typedef __attribute__((ext_vector_type(8)))  __bf16 v8bf;
typedef __attribute__((ext_vector_type(8)))  float  v8f;

#define T_STEPS 512
#define B_ROWS  1024
#define NWAVES  4           // 4 waves cooperate on one 16-row batch tile (one hidden tile each)
#define BLOCK_T (NWAVES*32)

__device__ __forceinline__ float fast_exp2(float x) { return __builtin_amdgcn_exp2f(x); }
__device__ __forceinline__ float fast_rcp (float x) { return __builtin_amdgcn_rcpf(x); }
__device__ __forceinline__ float sigm(float x) {
    return fast_rcp(1.0f + fast_exp2(-1.4426950408889634f * x));
}
__device__ __forceinline__ float tanh_fast(float x) {
    return 1.0f - 2.0f * fast_rcp(fast_exp2(2.8853900817779268f * x) + 1.0f);
}
__device__ __forceinline__ v8f splat8(float x) { return (v8f){x,x,x,x,x,x,x,x}; }

// Load a 16x32 bf16 A-fragment for this lane from an LDS row (row-major K).
// lane<16: M=lane,   K = {kb..kb+7, kb+16..kb+23}
// lane>=16:M=lane-16,K = {kb+8..kb+15, kb+24..kb+31}
__device__ __forceinline__ v16bf ld_afrag(const __bf16* row, int kbase) {
    v8bf lo = *(const v8bf*)(row + kbase);
    v8bf hi = *(const v8bf*)(row + kbase + 16);
    return __builtin_shufflevector(lo, hi, 0,1,2,3,4,5,6,7,8,9,10,11,12,13,14,15);
}

__global__ __launch_bounds__(BLOCK_T)
void lstm2_fused_kernel(const float* __restrict__ x,
                        const float* __restrict__ Wih0, const float* __restrict__ Whh0,
                        const float* __restrict__ bih0, const float* __restrict__ bhh0,
                        const float* __restrict__ Wih1, const float* __restrict__ Whh1,
                        const float* __restrict__ bih1, const float* __restrict__ bhh1,
                        const float* __restrict__ W1,   const float* __restrict__ b1,
                        const float* __restrict__ W2,   const float* __restrict__ b2,
                        float* __restrict__ out)
{
    // ---- LDS ----
    // Layer0 weights (augmented K=96): k<64 -> Whh0[col][k]; 64..69 -> Wih0[col][k-64]; else 0
    __shared__ alignas(32) __bf16 wf0[16*3*512];          // 48 KB
    // Layer1 weights (augmented K=128): k<64 -> Wih1[col][k]; else Whh1[col][k-64]
    __shared__ alignas(32) __bf16 wf1[16*4*512];          // 64 KB
    // Double-buffered activation staging (block-shared, 16-row tile)
    __shared__ alignas(32) __bf16 h0buf[2][16][64];       // 4 KB
    __shared__ alignas(32) __bf16 h1buf[2][16][64];       // 4 KB
    __shared__ alignas(32) __bf16 xbuf [2][16][32];       // 2 KB (x in K 0..5, zeros elsewhere)
    __shared__ float bias0l[256];
    __shared__ float bias1l[256];

    const int tid = threadIdx.x;

    // ---- one-time: bias combine, staging zero-init, weight swizzle to B-fragment layout ----
    for (int i = tid; i < 256; i += BLOCK_T) {
        bias0l[i] = bih0[i] + bhh0[i];
        bias1l[i] = bih1[i] + bhh1[i];
    }
    for (int i = tid; i < 2*16*64; i += BLOCK_T) { ((__bf16*)h0buf)[i] = (__bf16)0.0f; ((__bf16*)h1buf)[i] = (__bf16)0.0f; }
    for (int i = tid; i < 2*16*32; i += BLOCK_T) ((__bf16*)xbuf)[i] = (__bf16)0.0f;

    for (int e = tid; e < 16*3*512; e += BLOCK_T) {
        int hh = e & 15, ln = (e >> 4) & 31, fid = e >> 9;
        int kf = fid % 3, ct = fid / 3;
        int col = (ct >> 2) * 64 + (ct & 3) * 16 + (ln & 15);   // gate*64 + nt*16 + n
        int k   = kf * 32 + ((ln & 16) ? 8 : 0) + ((hh < 8) ? hh : hh + 8);
        float v = (k < 64) ? Whh0[col * 64 + k]
                : (k < 70) ? Wih0[col * 6 + (k - 64)]
                : 0.0f;
        wf0[e] = (__bf16)v;
    }
    for (int e = tid; e < 16*4*512; e += BLOCK_T) {
        int hh = e & 15, ln = (e >> 4) & 31, fid = e >> 9;
        int kf = fid & 3, ct = fid >> 2;
        int col = (ct >> 2) * 64 + (ct & 3) * 16 + (ln & 15);
        int k   = kf * 32 + ((ln & 16) ? 8 : 0) + ((hh < 8) ? hh : hh + 8);
        float v = (k < 64) ? Wih1[col * 64 + k] : Whh1[col * 64 + (k - 64)];
        wf1[e] = (__bf16)v;
    }
    __syncthreads();

    // ---- per-wave setup: wave wv owns hidden tile nt = wv (cols g*64 + wv*16 + n) ----
    const int wv   = tid >> 5;
    const int lane = tid & 31;
    const int r0   = blockIdx.x * 16;            // one 16-row batch tile per block
    const int M    = lane & 15;
    const int koff = (lane & 16) ? 8 : 0;        // A-frag K sub-offset
    const int n15  = lane & 15;
    const int mhi  = (lane & 16) ? 8 : 0;        // D-layout M offset

    // seed x_0 into xbuf[0]
    if (wv == 0 && lane < 16) {
        const float* xp = x + (size_t)(r0 + lane) * T_STEPS * 6;
        #pragma unroll
        for (int d = 0; d < 6; ++d) xbuf[0][lane][d] = (__bf16)xp[d];
    }

    float b0r[4], b1r[4];
    #pragma unroll
    for (int g = 0; g < 4; ++g) {
        int col = g * 64 + wv * 16 + n15;
        b0r[g] = bias0l[col];
        b1r[g] = bias1l[col];
    }
    v8f c0s = splat8(0.0f), c1s = splat8(0.0f);

    __syncthreads();

    // ---- sequential recurrence: both layers per step, 4 waves cooperating ----
    for (int t = 0; t < T_STEPS; ++t) {
        const int b = t & 1, p = b ^ 1;

        // prefetch x_{t+1} into the other x buffer (hidden behind this step's compute)
        if (wv == 0 && lane < 16 && (t + 1) < T_STEPS) {
            const float* xp = x + ((size_t)(r0 + lane) * T_STEPS + (t + 1)) * 6;
            #pragma unroll
            for (int d = 0; d < 6; ++d) xbuf[p][lane][d] = (__bf16)xp[d];
        }

        // ---- layer 0: gates(:,ct) = [h0_{t-1} | x_t | 0] @ W0aug + bias ----
        v16bf a0[3];
        a0[0] = ld_afrag(&h0buf[p][M][0], 0 + koff);
        a0[1] = ld_afrag(&h0buf[p][M][0], 32 + koff);
        a0[2] = ld_afrag(&xbuf[b][M][0], koff);

        v16bf wb0[12];
        #pragma unroll
        for (int g = 0; g < 4; ++g) {
            int ct = g * 4 + wv;
            #pragma unroll
            for (int kf = 0; kf < 3; ++kf)
                wb0[g * 3 + kf] = *(const v16bf*)&wf0[(ct * 3 + kf) * 512 + lane * 16];
        }

        v8f acc0[4];
        #pragma unroll
        for (int g = 0; g < 4; ++g) {
            v8f a = splat8(b0r[g]);
            #pragma unroll
            for (int kf = 0; kf < 3; ++kf)
                a = __builtin_amdgcn_wmma_f32_16x16x32_bf16(
                        false, a0[kf], false, wb0[g * 3 + kf], (short)0, a, false, false);
            acc0[g] = a;
        }
        #pragma unroll
        for (int e = 0; e < 8; ++e) {
            float ig = sigm(acc0[0][e]);
            float fg = sigm(acc0[1][e]);
            float gg = tanh_fast(acc0[2][e]);
            float og = sigm(acc0[3][e]);
            float c  = fg * c0s[e] + ig * gg;
            c0s[e] = c;
            h0buf[b][e + mhi][wv * 16 + n15] = (__bf16)(og * tanh_fast(c));
        }
        __syncthreads();   // h0_t visible to all waves

        // ---- layer 1: gates(:,ct) = [h0_t | h1_{t-1}] @ W1aug + bias ----
        v16bf a1[4];
        a1[0] = ld_afrag(&h0buf[b][M][0], 0 + koff);
        a1[1] = ld_afrag(&h0buf[b][M][0], 32 + koff);
        a1[2] = ld_afrag(&h1buf[p][M][0], 0 + koff);
        a1[3] = ld_afrag(&h1buf[p][M][0], 32 + koff);

        v16bf wb1[16];
        #pragma unroll
        for (int g = 0; g < 4; ++g) {
            int ct = g * 4 + wv;
            #pragma unroll
            for (int kf = 0; kf < 4; ++kf)
                wb1[g * 4 + kf] = *(const v16bf*)&wf1[(ct * 4 + kf) * 512 + lane * 16];
        }

        v8f acc1[4];
        #pragma unroll
        for (int g = 0; g < 4; ++g) {
            v8f a = splat8(b1r[g]);
            #pragma unroll
            for (int kf = 0; kf < 4; ++kf)
                a = __builtin_amdgcn_wmma_f32_16x16x32_bf16(
                        false, a1[kf], false, wb1[g * 4 + kf], (short)0, a, false, false);
            acc1[g] = a;
        }
        #pragma unroll
        for (int e = 0; e < 8; ++e) {
            float ig = sigm(acc1[0][e]);
            float fg = sigm(acc1[1][e]);
            float gg = tanh_fast(acc1[2][e]);
            float og = sigm(acc1[3][e]);
            float c  = fg * c1s[e] + ig * gg;
            c1s[e] = c;
            h1buf[b][e + mhi][wv * 16 + n15] = (__bf16)(og * tanh_fast(c));
        }
        __syncthreads();   // h1_t and x_{t+1} visible for next step
    }

    // ---- MLP head on final h1: relu(h@W1^T+b1)@W2^T+b2 -> sigmoid ----
    if (wv == 0 && lane < 16) {
        const __bf16* hrow = &h1buf[(T_STEPS - 1) & 1][lane][0];
        float hv[64];
        #pragma unroll
        for (int j = 0; j < 64; ++j) hv[j] = (float)hrow[j];
        float o = b2[0];
        for (int j = 0; j < 32; ++j) {
            float z = b1[j];
            #pragma unroll
            for (int k = 0; k < 64; ++k) z += hv[k] * W1[j * 64 + k];
            z = fmaxf(z, 0.0f);
            o += z * W2[j];
        }
        out[r0 + lane] = sigm(o);
    }
}

extern "C" void kernel_launch(void* const* d_in, const int* in_sizes, int n_in,
                              void* d_out, int out_size, void* d_ws, size_t ws_size,
                              hipStream_t stream) {
    (void)in_sizes; (void)n_in; (void)d_ws; (void)ws_size; (void)out_size;
    const float* x    = (const float*)d_in[0];
    const float* Wih0 = (const float*)d_in[1];
    const float* Whh0 = (const float*)d_in[2];
    const float* bih0 = (const float*)d_in[3];
    const float* bhh0 = (const float*)d_in[4];
    const float* Wih1 = (const float*)d_in[5];
    const float* Whh1 = (const float*)d_in[6];
    const float* bih1 = (const float*)d_in[7];
    const float* bhh1 = (const float*)d_in[8];
    const float* W1   = (const float*)d_in[9];
    const float* b1   = (const float*)d_in[10];
    const float* W2   = (const float*)d_in[11];
    const float* b2   = (const float*)d_in[12];
    float* out = (float*)d_out;

    dim3 grid(B_ROWS / 16);   // 64 blocks, one 16-row batch tile each
    dim3 block(BLOCK_T);      // 4 waves cooperating per tile
    lstm2_fused_kernel<<<grid, block, 0, stream>>>(
        x, Wih0, Whh0, bih0, bhh0, Wih1, Whh1, bih1, bhh1, W1, b1, W2, b2, out);
}
